// TAG_47459388621620
// MI455X (gfx1250) — matrix operand order repaired and checked by reading, not compiled, not run on hardware
//
#include <hip/hip_runtime.h>

// ---------------- types for WMMA fragments (wave32, gfx1250) ----------------
typedef __bf16 v16bf __attribute__((ext_vector_type(16)));
typedef float  v8f   __attribute__((ext_vector_type(8)));
typedef unsigned int v4u __attribute__((ext_vector_type(4)));

#define DD    240      // feature dim
#define RREL  8        // edge relations
#define NRELW 9        // 8 relations + root folded in as r==8
#define HBS   256      // padded K stride for bf16 activations
#define WTS   256      // padded K stride for bf16 weights
#define NTW   5        // N-tiles per wave (5*16 = 80 output cols)
#define NGRP  3        // groups of N-tiles: 3*5*16 = 240
#define KSTEPS (HBS / 32)

__device__ __forceinline__ unsigned short f2bf(float f) {
  unsigned int u = __float_as_uint(f);
  unsigned int r = u + 0x7FFFu + ((u >> 16) & 1u);   // round-to-nearest-even
  return (unsigned short)(r >> 16);
}

__device__ __forceinline__ float lrelu(float v) { return v > 0.f ? v : 0.01f * v; }

// ---------------- 1) build per-relation weights, transposed + bf16 ----------
// wt[r][o][i] = sum_b comp[r,b]*basis[b,i,o]  (r<8),  root[i,o] (r==8)
// stored row-major over (r,o) with i-stride WTS (zero padded 240..255)
__global__ void wt_prep_kernel(const float* __restrict__ comp,
                               const float* __restrict__ basis,
                               const float* __restrict__ root,
                               unsigned short* __restrict__ wt, int layer) {
  const float* compL  = comp  + (size_t)layer * RREL * 4;
  const float* basisL = basis + (size_t)layer * 4 * DD * DD;
  const float* rootL  = root  + (size_t)layer * DD * DD;
  const int total = NRELW * DD * WTS;
  for (int i = blockIdx.x * blockDim.x + threadIdx.x; i < total;
       i += gridDim.x * blockDim.x) {
    int k = i % WTS;              // input dim
    int o = (i / WTS) % DD;       // output dim
    int r = i / (WTS * DD);
    float v = 0.f;
    if (k < DD) {
      if (r < RREL) {
        float a = 0.f;
#pragma unroll
        for (int b = 0; b < 4; ++b)
          a += compL[r * 4 + b] * basisL[((size_t)b * DD + k) * DD + o];
        v = a;
      } else {
        v = rootL[(size_t)k * DD + o];
      }
    }
    wt[i] = f2bf(v);
  }
}

// ---------------- 2) f32 activations -> padded bf16 -------------------------
__global__ void x_to_bf_kernel(const float* __restrict__ x,
                               unsigned short* __restrict__ hbf, int n_nodes) {
  const long long total = (long long)n_nodes * HBS;
  for (long long i = blockIdx.x * (long long)blockDim.x + threadIdx.x; i < total;
       i += (long long)gridDim.x * blockDim.x) {
    int c = (int)(i % HBS);
    long long n = i / HBS;
    float v = (c < DD) ? x[n * DD + c] : 0.f;
    hbf[i] = f2bf(v);
  }
}

// ---------------- 3) WMMA GEMM: xw[n][r][:] = h[n] @ W_r ---------------------
// one wave per (16-node tile, relation, 5-wide N-group). r==8 -> agg (root).
__global__ __launch_bounds__(256, 1)
void rgcn_gemm_kernel(const unsigned short* __restrict__ hbf,
                      const unsigned short* __restrict__ wt,
                      float* __restrict__ xw,
                      float* __restrict__ agg, int m_tiles) {
  const int wave = blockIdx.x * (blockDim.x >> 5) + (threadIdx.x >> 5);
  const int lane = threadIdx.x & 31;
  const int total = m_tiles * NRELW * NGRP;
  if (wave >= total) return;
  const int ng = wave % NGRP;            // N-group 0..2
  const int r  = (wave / NGRP) % NRELW;  // relation (8 == root)
  const int mt = wave / (NGRP * NRELW);  // 16-row node tile

  const int m  = lane & 15;       // A row within tile / B,N column
  const int kh = lane >> 4;       // K-half select

  v8f acc[NTW];
#pragma unroll
  for (int t = 0; t < NTW; ++t)
#pragma unroll
    for (int j = 0; j < 8; ++j) acc[t][j] = 0.f;

  union Frag { v16bf v; v4u q[2]; };

  const unsigned short* arow  = hbf + ((size_t)(mt * 16 + m)) * HBS + kh * 8;
  const unsigned short* wbase = wt + (size_t)r * DD * WTS +
                                (size_t)(ng * NTW * 16 + m) * WTS + kh * 8;

#pragma unroll
  for (int k0 = 0; k0 < HBS; k0 += 32) {
    Frag a;
    a.q[0] = *(const v4u*)(arow + k0);        // K = k0 + kh*8 + 0..7
    a.q[1] = *(const v4u*)(arow + k0 + 16);   // K = k0 + 16 + kh*8 + 0..7
#pragma unroll
    for (int nt = 0; nt < NTW; ++nt) {
      Frag b;
      const unsigned short* bp = wbase + (size_t)nt * 16 * WTS + k0;
      b.q[0] = *(const v4u*)(bp);             // K = k0 + kh*8 + 0..7
      b.q[1] = *(const v4u*)(bp + 16);        // K = k0 + 16 + kh*8 + 0..7
      acc[nt] = __builtin_amdgcn_wmma_f32_16x16x32_bf16(
          false, a.v, false, b.v, (short)0, acc[nt], false, false);
    }
  }

  // C/D layout: lane 0-15 -> N=lane, M=j ; lane 16-31 -> N=lane-16, M=8+j
  const int n = lane & 15;
  const int mbase = mt * 16 + kh * 8;
  const int cbase = ng * NTW * 16 + n;
  if (r < RREL) {
#pragma unroll
    for (int nt = 0; nt < NTW; ++nt)
#pragma unroll
      for (int j = 0; j < 8; ++j)
        xw[((size_t)(mbase + j) * RREL + r) * DD + cbase + nt * 16] = acc[nt][j];
  } else {
#pragma unroll
    for (int nt = 0; nt < NTW; ++nt)
#pragma unroll
      for (int j = 0; j < 8; ++j)
        agg[(size_t)(mbase + j) * DD + cbase + nt * 16] = acc[nt][j];
  }
}

// ---------------- 4) edge gather + scatter-add (f32 atomics) -----------------
// 64 lanes per edge, lanes 0..59 each move one float4 of the 240-f32 message.
__global__ void edge_agg_kernel(const float* __restrict__ xw,
                                const int* __restrict__ src,
                                const int* __restrict__ dst,
                                const int* __restrict__ et,
                                float* __restrict__ agg, int n_edges) {
  const int epb = blockDim.x >> 6;          // edges per block
  const int sub = threadIdx.x >> 6;
  const int q   = threadIdx.x & 63;
  for (long long e0 = (long long)blockIdx.x * epb; e0 < n_edges;
       e0 += (long long)gridDim.x * epb) {
    long long e = e0 + sub;
    if (e < n_edges && q < 60) {
      int s = src[e], d = dst[e], t = et[e];
      const float4 v = *(const float4*)(xw + ((size_t)s * RREL + t) * DD + (size_t)q * 4);
      float* ap = agg + (size_t)d * DD + (size_t)q * 4;
      atomicAdd(ap + 0, v.x);
      atomicAdd(ap + 1, v.y);
      atomicAdd(ap + 2, v.z);
      atomicAdd(ap + 3, v.w);
    }
  }
}

// ---------------- 5) bias + LeakyReLU + bf16 (inter-layer epilogue) ----------
__global__ void finish_layer_kernel(const float* __restrict__ agg,
                                    const float* __restrict__ bias,
                                    unsigned short* __restrict__ hbf, int n_nodes) {
  const long long total = (long long)n_nodes * HBS;
  for (long long i = blockIdx.x * (long long)blockDim.x + threadIdx.x; i < total;
       i += (long long)gridDim.x * blockDim.x) {
    int c = (int)(i % HBS);
    long long n = i / HBS;
    float v = 0.f;
    if (c < DD) v = lrelu(agg[n * DD + c] + bias[c]);
    hbf[i] = f2bf(v);
  }
}

// ---------------- 6) MLP head on two node embeddings -------------------------
__global__ void head_kernel(const float* __restrict__ agg,
                            const float* __restrict__ bias,
                            const float* __restrict__ W1, const float* __restrict__ b1,
                            const float* __restrict__ W2, const float* __restrict__ b2,
                            const int* __restrict__ cip, const int* __restrict__ sip,
                            float* __restrict__ out) {
  __shared__ float feat[2 * DD];
  __shared__ float red[128];
  const int tid = threadIdx.x;
  const int ci = cip[0], si = sip[0];
  for (int c = tid; c < DD; c += blockDim.x) {
    float hs = agg[(size_t)si * DD + c] + bias[c];
    float hc = agg[(size_t)ci * DD + c] + bias[c];
    feat[c]      = fabsf(hs - hc);
    feat[DD + c] = hs * hc;
  }
  __syncthreads();
  float a = b1[tid];
  for (int c = 0; c < 2 * DD; ++c) a += W1[(size_t)tid * 2 * DD + c] * feat[c];
  red[tid] = lrelu(a) * W2[tid];
  __syncthreads();
  for (int s = 64; s > 0; s >>= 1) {
    if (tid < s) red[tid] += red[tid + s];
    __syncthreads();
  }
  if (tid == 0) out[0] = red[0] + b2[0];
}

// ---------------- launcher ---------------------------------------------------
extern "C" void kernel_launch(void* const* d_in, const int* in_sizes, int n_in,
                              void* d_out, int out_size, void* d_ws, size_t ws_size,
                              hipStream_t stream) {
  const float* x     = (const float*)d_in[0];
  const int*   ei    = (const int*)d_in[1];   // [2, E]
  const int*   et    = (const int*)d_in[2];   // [E]
  const float* comp  = (const float*)d_in[3]; // [L, R, B]
  const float* basis = (const float*)d_in[4]; // [L, B, D, D]
  const float* root  = (const float*)d_in[5]; // [L, D, D]
  const float* bias  = (const float*)d_in[6]; // [L, D]
  const float* W1    = (const float*)d_in[7];
  const float* b1    = (const float*)d_in[8];
  const float* W2    = (const float*)d_in[9];
  const float* b2    = (const float*)d_in[10];
  const int*   cip   = (const int*)d_in[11];
  const int*   sip   = (const int*)d_in[12];
  float* out = (float*)d_out;

  const int nNodes = in_sizes[0] / DD;        // 50000
  const int nEdges = in_sizes[1] / 2;         // 1.6M
  const int mTiles = nNodes / 16;             // 3125

  // workspace carve-out (all 16B-aligned)
  char* ws = (char*)d_ws;
  size_t off = 0;
  float* xw = (float*)(ws + off);            off += (size_t)nNodes * RREL * DD * sizeof(float);
  float* agg = (float*)(ws + off);           off += (size_t)nNodes * DD * sizeof(float);
  unsigned short* hbf = (unsigned short*)(ws + off); off += (size_t)nNodes * HBS * sizeof(unsigned short);
  unsigned short* wt  = (unsigned short*)(ws + off); off += (size_t)NRELW * DD * WTS * sizeof(unsigned short);
  (void)ws_size; (void)n_in; (void)out_size;

  const int* srcp = ei;
  const int* dstp = ei + nEdges;

  const int gemmWaves  = mTiles * NRELW * NGRP;
  const int gemmBlocks = (gemmWaves + 7) / 8;          // 8 waves / 256-thread block

  // ---- layer 0 ----
  wt_prep_kernel<<<256, 256, 0, stream>>>(comp, basis, root, wt, 0);
  x_to_bf_kernel<<<2048, 256, 0, stream>>>(x, hbf, nNodes);
  rgcn_gemm_kernel<<<gemmBlocks, 256, 0, stream>>>(hbf, wt, xw, agg, mTiles);
  edge_agg_kernel<<<32768, 256, 0, stream>>>(xw, srcp, dstp, et, agg, nEdges);
  finish_layer_kernel<<<2048, 256, 0, stream>>>(agg, bias + 0 * DD, hbf, nNodes);

  // ---- layer 1 ----
  wt_prep_kernel<<<256, 256, 0, stream>>>(comp, basis, root, wt, 1);
  rgcn_gemm_kernel<<<gemmBlocks, 256, 0, stream>>>(hbf, wt, xw, agg, mTiles);
  edge_agg_kernel<<<32768, 256, 0, stream>>>(xw, srcp, dstp, et, agg, nEdges);

  // ---- head ----
  head_kernel<<<1, 128, 0, stream>>>(agg, bias + 1 * DD, W1, b1, W2, b2, cip, sip, out);
}
